// PairformerStack_17952963297595
// MI455X (gfx1250) — matrix operand order repaired and checked by reading, not compiled, not run on hardware
//
#include <hip/hip_runtime.h>
#include <hip/hip_bf16.h>
#include <cstdint>

// ---------------------------------------------------------------------------
// Pairformer stack for gfx1250 (MI455X). bf16 WMMA GEMM core + fp32 epilogues.
// All matmuls (projections, triangle einsum, attention QK^T / PV) run through
// one vectorized 128x64x32-tile WMMA kernel with async global->LDS staging.
// Host-guaranteed: csA==csB==1, K%32==0, M%128==0; transB implies N%64==0.
// ---------------------------------------------------------------------------

typedef __attribute__((ext_vector_type(16))) __bf16 bf16x16;
typedef __attribute__((ext_vector_type(8)))  float  f32x8;

__device__ __forceinline__ unsigned short f2bf(float f) {
  union { float f; unsigned u; } c; c.f = f;
  unsigned u = c.u;
  u += 0x7FFFu + ((u >> 16) & 1u);   // round-to-nearest-even
  return (unsigned short)(u >> 16);
}
__device__ __forceinline__ float sigm(float x) { return 1.f / (1.f + __expf(-x)); }

// CDNA5 async global->LDS copy (ASYNCcnt-tracked), 16B per lane.
// LDS generic address: low 32 bits carry the LDS byte offset (ISA aperture rule).
__device__ __forceinline__ void async_copy16(const unsigned short* g, unsigned short* l) {
  unsigned lds = (unsigned)(unsigned long long)l;
  asm volatile("global_load_async_to_lds_b128 %0, %1, off"
               :: "v"(lds), "v"(g) : "memory");
}
__device__ __forceinline__ void async_wait0() {
  asm volatile("s_wait_asynccnt 0" ::: "memory");
}

struct GemmDesc {
  int M, N, K;
  long long rsA, b0A, b1A;
  long long rsB, b0B, b1B;
  long long rsC, csC, b0C, b1C;
  int bmod, transB;
  float alpha;
};

#define TM 128
#define TN 64
#define TK 32
#define LDP 40   // padded LDS pitch (elements); rows stay 16B-aligned

union Frag { bf16x16 v; uint4 q[2]; };

__global__ __launch_bounds__(256)
void k_gemm_v(const unsigned short* __restrict__ A,
              const unsigned short* __restrict__ B,
              float* __restrict__ C, GemmDesc d)
{
  __shared__ unsigned short sA[TM * LDP];
  __shared__ unsigned short sB[TN * LDP];

  const int tid = threadIdx.x, lane = tid & 31, wave = tid >> 5;
  const int m0 = blockIdx.x * TM, n0 = blockIdx.y * TN, bz = blockIdx.z;
  const long long offA = (long long)(bz % d.bmod) * d.b0A + (long long)(bz / d.bmod) * d.b1A;
  const long long offB = (long long)(bz % d.bmod) * d.b0B + (long long)(bz / d.bmod) * d.b1B;
  const long long offC = (long long)(bz % d.bmod) * d.b0C + (long long)(bz / d.bmod) * d.b1C;
  const int wm = wave & 3, wn = wave >> 2, lm = lane & 15, lh = lane >> 4;

  f32x8 acc[2][2] = {};

  for (int k0 = 0; k0 < d.K; k0 += TK) {
    // ---- A tile: 128x32, 2 async 16B chunks per thread (M%128==0) ----
#pragma unroll
    for (int r = 0; r < 2; ++r) {
      int row = (tid >> 2) + r * 64;
      int kq  = (tid & 3) * 8;
      async_copy16(A + offA + (long long)(m0 + row) * d.rsA + (k0 + kq),
                   &sA[row * LDP + kq]);
    }
    // ---- B tile -> LDS [n][k] ----
    if (d.transB) {                    // B stored [N][K], N%64==0: async copy
      int n  = tid >> 2;
      int kq = (tid & 3) * 8;
      async_copy16(B + offB + (long long)(n0 + n) * d.rsB + (k0 + kq),
                   &sB[n * LDP + kq]);
    } else {                           // B stored [K][N]: vector load + transpose store
      int k  = tid >> 3;
      int nq = (tid & 7) * 8;
      long long rb = offB + (long long)(k0 + k) * d.rsB;
      unsigned short tmp[8];
      if (n0 + nq + 8 <= d.N) {
        *(uint4*)tmp = *(const uint4*)(B + rb + (n0 + nq));
      } else {
#pragma unroll
        for (int e = 0; e < 8; ++e) {
          int gn = n0 + nq + e;
          unsigned short v = B[rb + (gn < d.N ? gn : d.N - 1)];
          tmp[e] = (gn < d.N) ? v : (unsigned short)0;
        }
      }
#pragma unroll
      for (int e = 0; e < 8; ++e) sB[(nq + e) * LDP + k] = tmp[e];
    }
    async_wait0();
    __syncthreads();

    Frag fa[2], fb[2];
#pragma unroll
    for (int i = 0; i < 2; ++i) {
      // A (ISA 16-bit A 16x32): lane m, K = lh*8+0..7 then 16+lh*8+0..7
      const unsigned short* pa = &sA[(wm * 32 + i * 16 + lm) * LDP + lh * 8];
      fa[i].q[0] = *(const uint4*)pa;
      fa[i].q[1] = *(const uint4*)(pa + 16);
      // B: lane n, K = lh*8+0..7 then 16+lh*8+0..7
      const unsigned short* pb = &sB[(wn * 32 + i * 16 + lm) * LDP + lh * 8];
      fb[i].q[0] = *(const uint4*)pb;
      fb[i].q[1] = *(const uint4*)(pb + 16);
    }
#pragma unroll
    for (int i = 0; i < 2; ++i)
#pragma unroll
      for (int j = 0; j < 2; ++j)
        acc[i][j] = __builtin_amdgcn_wmma_f32_16x16x32_bf16(
            false, fa[i].v, false, fb[j].v, (short)0, acc[i][j], false, false);
    __syncthreads();
  }

  // ---- store (full M; guard N only) ----
#pragma unroll
  for (int i = 0; i < 2; ++i)
#pragma unroll
    for (int j = 0; j < 2; ++j)
#pragma unroll
      for (int r = 0; r < 8; ++r) {
        int gm = m0 + wm * 32 + i * 16 + lh * 8 + r;   // C: VGPR r -> M=r / 8+r
        int gn = n0 + wn * 32 + j * 16 + lm;
        if (gn < d.N)
          C[offC + (long long)gm * d.rsC + (long long)gn * d.csC] = d.alpha * acc[i][j][r];
      }
}

// ---------------------------------------------------------------------------
// LayerNorm (wave per row), f32 in -> bf16 out
// ---------------------------------------------------------------------------
__global__ __launch_bounds__(256)
void k_layernorm(const float* __restrict__ x, const float* __restrict__ g,
                 const float* __restrict__ b, unsigned short* __restrict__ out,
                 long long rows, int C)
{
  long long row = (long long)blockIdx.x * 8 + (threadIdx.x >> 5);
  int lane = threadIdx.x & 31;
  if (row >= rows) return;
  const float* xr = x + row * C;
  float sum = 0.f, sq = 0.f;
  for (int c = lane; c < C; c += 32) { float v = xr[c]; sum += v; sq += v * v; }
  for (int o = 16; o; o >>= 1) { sum += __shfl_xor(sum, o, 32); sq += __shfl_xor(sq, o, 32); }
  float mu = sum / C;
  float rinv = rsqrtf(sq / C - mu * mu + 1e-5f);
  unsigned short* orow = out + row * C;
  for (int c = lane; c < C; c += 32)
    orow[c] = f2bf((xr[c] - mu) * rinv * g[c] + b[c]);
}

// ---------------------------------------------------------------------------
// Elementwise kernels
// ---------------------------------------------------------------------------
__global__ void k_to_bf16(const float* __restrict__ in, unsigned short* __restrict__ out, long long n) {
  long long i = (long long)blockIdx.x * 256 + threadIdx.x;
  if (i < n) out[i] = f2bf(in[i]);
}
// apb q/k: repack f32 [256, 16*24] -> bf16 [256, 16, 32] zero-padded (24->32),
// optional per-channel bias, so QK^T runs K=32 on the fast GEMM path.
__global__ void k_pad_head(const float* __restrict__ in, const float* __restrict__ bias,
                           unsigned short* __restrict__ out, int useBias) {
  int i = blockIdx.x * 256 + threadIdx.x;
  if (i >= 256 * 16 * 32) return;
  int dd = i & 31, h = (i >> 5) & 15, r = i >> 9;
  float v = 0.f;
  if (dd < 24) {
    int c = h * 24 + dd;
    v = in[r * 384 + c] + (useBias ? bias[c] : 0.f);
  }
  out[i] = f2bf(v);
}
// Gated masked projection for tri-mul, written PACKED as [c][256][256] so the
// triangle einsum runs as contiguous batched GEMM. xpose selects [c][i][k]
// (outgoing) vs transposed-within-channel (incoming).
__global__ void k_gate_pack(const float* __restrict__ G, const float* __restrict__ P,
                            const float* __restrict__ mask, unsigned short* __restrict__ out,
                            int xpose) {
  long long n = 65536LL * 128;
  long long i = (long long)blockIdx.x * 256 + threadIdx.x;
  if (i >= n) return;
  long long r = i >> 7;            // pair row (i*256+k)
  int c = (int)(i & 127);
  int rhi = (int)(r >> 8), rlo = (int)(r & 255);
  float v = mask[r] * sigm(G[i]) * P[i];
  long long o = ((long long)c << 16) + (xpose ? ((long long)rlo << 8) + rhi : r);
  out[o] = f2bf(v);
}
__global__ void k_gate_residual(float* __restrict__ x, const float* __restrict__ g,
                                const float* __restrict__ o, long long n) {
  long long i = (long long)blockIdx.x * 256 + threadIdx.x;
  if (i < n) x[i] += sigm(g[i]) * o[i];
}
__global__ void k_silu_mul(const float* __restrict__ a, const float* __restrict__ b,
                           unsigned short* __restrict__ out, long long n) {
  long long i = (long long)blockIdx.x * 256 + threadIdx.x;
  if (i < n) { float v = a[i]; out[i] = f2bf(v * sigm(v) * b[i]); }
}
__global__ void k_add(float* __restrict__ x, const float* __restrict__ y, long long n) {
  long long i = (long long)blockIdx.x * 256 + threadIdx.x;
  if (i < n) x[i] += y[i];
}
__global__ void k_transpose(const float* __restrict__ in, float* __restrict__ out, int N, int C) {
  long long total = (long long)N * N * C;
  long long i = (long long)blockIdx.x * 256 + threadIdx.x;
  if (i >= total) return;
  int c = (int)(i % C);
  int jj = (int)((i / C) % N);
  int ii = (int)(i / ((long long)C * N));
  out[((long long)jj * N + ii) * C + c] = in[i];
}

// ---------------------------------------------------------------------------
// Softmax: triangle attention. scores layout [ic][h(4)][j(256)][k(256)]
// ---------------------------------------------------------------------------
__global__ __launch_bounds__(256)
void k_softmax_tri(float* __restrict__ sc, const float* __restrict__ tb,
                   const float* __restrict__ mask, unsigned short* __restrict__ probs,
                   int i0, int nrows)
{
  int row = blockIdx.x * 8 + (threadIdx.x >> 5);
  int lane = threadIdx.x & 31;
  if (row >= nrows) return;
  int j = row & 255, h = (row >> 8) & 3, ic = row >> 10;
  int i = i0 + ic;
  const float* r = sc + (long long)row * 256;
  float v[8], mx = -3.0e38f;
#pragma unroll
  for (int t = 0; t < 8; ++t) {
    int k = t * 32 + lane;
    float x = r[k] + tb[(((long long)j << 8) + k) * 4 + h] + (mask[((long long)i << 8) + k] - 1.f) * 1e9f;
    v[t] = x; mx = fmaxf(mx, x);
  }
  for (int o = 16; o; o >>= 1) mx = fmaxf(mx, __shfl_xor(mx, o, 32));
  float s = 0.f;
#pragma unroll
  for (int t = 0; t < 8; ++t) { v[t] = __expf(v[t] - mx); s += v[t]; }
  for (int o = 16; o; o >>= 1) s += __shfl_xor(s, o, 32);
  float inv = 1.f / s;
  unsigned short* pr = probs + (long long)row * 256;
#pragma unroll
  for (int t = 0; t < 8; ++t) pr[t * 32 + lane] = f2bf(v[t] * inv);
}

// AttentionPairBias softmax: scores [h(16)][i(256)][j(256)]; += zb[i,j,h]
__global__ __launch_bounds__(256)
void k_softmax_apb(float* __restrict__ sc, const float* __restrict__ zb,
                   unsigned short* __restrict__ probs, int nrows)
{
  int row = blockIdx.x * 8 + (threadIdx.x >> 5);
  int lane = threadIdx.x & 31;
  if (row >= nrows) return;
  int i = row & 255, h = row >> 8;
  const float* r = sc + (long long)row * 256;
  float v[8], mx = -3.0e38f;
#pragma unroll
  for (int t = 0; t < 8; ++t) {
    int j = t * 32 + lane;
    float x = r[j] + zb[(((long long)i << 8) + j) * 16 + h];
    v[t] = x; mx = fmaxf(mx, x);
  }
  for (int o = 16; o; o >>= 1) mx = fmaxf(mx, __shfl_xor(mx, o, 32));
  float s = 0.f;
#pragma unroll
  for (int t = 0; t < 8; ++t) { v[t] = __expf(v[t] - mx); s += v[t]; }
  for (int o = 16; o; o >>= 1) s += __shfl_xor(s, o, 32);
  float inv = 1.f / s;
  unsigned short* pr = probs + (long long)row * 256;
#pragma unroll
  for (int t = 0; t < 8; ++t) pr[t * 32 + lane] = f2bf(v[t] * inv);
}

// ---------------------------------------------------------------------------
// Host orchestration
// ---------------------------------------------------------------------------
static const int N_ = 256, CZ = 128, CS = 384, HP = 4, CA = 32, HS = 16;
static const long long RZ = (long long)N_ * N_;   // 65536 pair rows
static const int CI_CHUNK = 32;                    // attention i-chunk

extern "C" void kernel_launch(void* const* d_in, const int* in_sizes, int n_in,
                              void* d_out, int out_size, void* d_ws, size_t ws_size,
                              hipStream_t stream)
{
  (void)in_sizes; (void)n_in; (void)out_size; (void)ws_size;

  const float* mask = (const float*)d_in[2];
  const float* P[114];
  for (int i = 0; i < 114; ++i) P[i] = (const float*)d_in[3 + i];

  float* s = (float*)d_out;                        // [256, 384]
  float* z = (float*)d_out + (long long)N_ * CS;   // [256,256,128]
  hipMemcpyAsync(s, d_in[0], (size_t)N_ * CS * 4, hipMemcpyDeviceToDevice, stream);
  hipMemcpyAsync(z, d_in[1], (size_t)RZ * CZ * 4, hipMemcpyDeviceToDevice, stream);

  // ---- workspace carve ----
  char* wp = (char*)d_ws;
  auto carve = [&](size_t bytes) { char* r = wp; wp += (bytes + 255) & ~(size_t)255; return r; };
  float* t0   = (float*)carve(RZ * CZ * 4);
  float* t1   = (float*)carve(RZ * CZ * 4);
  float* scb  = (float*)carve((long long)CI_CHUNK * HP * N_ * N_ * 4);
  float* ztmp = (float*)carve(RZ * CZ * 4);
  unsigned short* bfx = (unsigned short*)carve(RZ * CZ * 2);
  unsigned short* bf0 = (unsigned short*)carve(RZ * CZ * 2);
  unsigned short* bf1 = (unsigned short*)carve(RZ * CZ * 2);
  unsigned short* bf2 = (unsigned short*)carve(RZ * CZ * 2);
  unsigned short* pbf = (unsigned short*)carve((long long)CI_CHUNK * HP * N_ * N_ * 2);
  float* tbb  = (float*)carve(RZ * HP * 4);
  float* zbb  = (float*)carve(RZ * HS * 4);
  unsigned short* w0 = (unsigned short*)carve(589824 * 2);
  unsigned short* w1 = (unsigned short*)carve(589824 * 2);
  unsigned short* w2 = (unsigned short*)carve(589824 * 2);
  unsigned short* sbf = (unsigned short*)carve((long long)N_ * CS * 2);
  unsigned short* qbf = (unsigned short*)carve((long long)N_ * HS * 32 * 2);  // padded q
  unsigned short* kbf = (unsigned short*)carve((long long)N_ * HS * 32 * 2);  // padded k
  unsigned short* vbf = (unsigned short*)carve((long long)N_ * CS * 2);
  unsigned short* obf = (unsigned short*)carve((long long)N_ * CS * 2);
  float* st0 = (float*)carve((long long)N_ * 1536 * 4);
  float* st1 = (float*)carve((long long)N_ * 1536 * 4);

  auto NB = [](long long n) { return (unsigned)((n + 255) / 256); };

  auto cvt = [&](const float* src, unsigned short* dst, long long n) {
    k_to_bf16<<<NB(n), 256, 0, stream>>>(src, dst, n);
  };
  auto ln = [&](const float* x, const float* g, const float* b,
                unsigned short* o, long long rows, int C) {
    k_layernorm<<<(unsigned)((rows + 7) / 8), 256, 0, stream>>>(x, g, b, o, rows, C);
  };
  auto gemm = [&](const unsigned short* A, const unsigned short* B, float* C,
                  int M, int Nc, int K,
                  long long rsA, long long b0A, long long b1A,
                  long long rsB, long long b0B, long long b1B,
                  long long rsC, long long csC, long long b0C, long long b1C,
                  int batch, int bmod, int transB, float alpha) {
    GemmDesc d{M, Nc, K, rsA, b0A, b1A, rsB, b0B, b1B,
               rsC, csC, b0C, b1C, bmod, transB, alpha};
    dim3 g((M + TM - 1) / TM, (Nc + TN - 1) / TN, batch);
    k_gemm_v<<<g, 256, 0, stream>>>(A, B, C, d);
  };
  auto gemm2 = [&](const unsigned short* A, const unsigned short* B, float* C,
                   int M, int Nc, int K) {  // plain row-major C = A[MxK] @ B[KxN]
    gemm(A, B, C, M, Nc, K, K, 0, 0, Nc, 0, 0, Nc, 1, 0, 0, 1, 1, 0, 1.f);
  };

  // ---- Triangle multiplicative update (in place on zb) ----
  auto tri_mul = [&](float* zb, const float* const* p, bool outgoing) {
    ln(zb, p[0], p[1], bfx, RZ, CZ);
    int xp = outgoing ? 0 : 1;
    cvt(p[2], w0, CZ * CZ); gemm2(bfx, w0, t0, (int)RZ, CZ, CZ);
    cvt(p[3], w1, CZ * CZ); gemm2(bfx, w1, t1, (int)RZ, CZ, CZ);
    k_gate_pack<<<NB(RZ * CZ), 256, 0, stream>>>(t0, t1, mask, bf0, xp);
    cvt(p[4], w0, CZ * CZ); gemm2(bfx, w0, t0, (int)RZ, CZ, CZ);
    cvt(p[5], w1, CZ * CZ); gemm2(bfx, w1, t1, (int)RZ, CZ, CZ);
    k_gate_pack<<<NB(RZ * CZ), 256, 0, stream>>>(t0, t1, mask, bf1, xp);
    // triangle einsum: 128 channel batches of [256x256] @ [256x256]^T
    gemm(bf0, bf1, t0, N_, N_, N_,
         N_, RZ, 0,               // A_c[i][k]
         N_, RZ, 0,               // B_c[j][k] (transB)
         (long long)N_ * CZ, CZ, 1, 0,
         CZ, CZ, 1, 1.f);
    ln(t0, p[7], p[8], bf2, RZ, CZ);
    cvt(p[9], w0, CZ * CZ); gemm2(bf2, w0, t1, (int)RZ, CZ, CZ);   // t @ out
    cvt(p[6], w1, CZ * CZ); gemm2(bfx, w1, t0, (int)RZ, CZ, CZ);   // gate
    k_gate_residual<<<NB(RZ * CZ), 256, 0, stream>>>(zb, t0, t1, RZ * CZ);
  };

  // ---- Triangle attention, starting node (in place on zb) ----
  auto tri_att = [&](float* zb, const float* const* p) {
    ln(zb, p[0], p[1], bfx, RZ, CZ);
    cvt(p[2], w0, CZ * CZ); gemm2(bfx, w0, t0, (int)RZ, CZ, CZ); cvt(t0, bf0, RZ * CZ); // q
    cvt(p[3], w0, CZ * CZ); gemm2(bfx, w0, t0, (int)RZ, CZ, CZ); cvt(t0, bf1, RZ * CZ); // k
    cvt(p[4], w0, CZ * CZ); gemm2(bfx, w0, t0, (int)RZ, CZ, CZ); cvt(t0, bf2, RZ * CZ); // v
    cvt(p[5], w0, CZ * HP); gemm2(bfx, w0, tbb, (int)RZ, HP, CZ);                        // tri bias
    const float scale = 0.17677669529663687f;  // 1/sqrt(32)
    const long long RI = (long long)N_ * CZ;   // stride per i
    for (int i0 = 0; i0 < N_; i0 += CI_CHUNK) {
      // scores[ic][h][j][k] = scale * q[i,j,h,:] . k[i,k,h,:]  (batch z = ic*4+h)
      gemm(bf0 + (long long)i0 * RI, bf1 + (long long)i0 * RI, scb,
           N_, N_, CA,
           CZ, CA, RI,
           CZ, CA, RI,
           N_, 1, (long long)N_ * N_, (long long)HP * N_ * N_,
           CI_CHUNK * HP, HP, 1, scale);
      int nrows = CI_CHUNK * HP * N_;
      k_softmax_tri<<<(unsigned)((nrows + 7) / 8), 256, 0, stream>>>(scb, tbb, mask, pbf, i0, nrows);
      // o[j, h*32+c] = sum_k P[j,k] v[i,k,h,c]
      gemm(pbf, bf2 + (long long)i0 * RI, t0 + (long long)i0 * RI,
           N_, CA, N_,
           N_, (long long)N_ * N_, (long long)HP * N_ * N_,
           CZ, CA, RI,
           CZ, 1, CA, RI,
           CI_CHUNK * HP, HP, 0, 1.f);
    }
    cvt(t0, bf0, RZ * CZ);                                         // concat heads
    cvt(p[7], w0, CZ * CZ); gemm2(bf0, w0, t1, (int)RZ, CZ, CZ);   // out proj
    cvt(p[6], w0, CZ * CZ); gemm2(bfx, w0, t0, (int)RZ, CZ, CZ);   // gate
    k_gate_residual<<<NB(RZ * CZ), 256, 0, stream>>>(zb, t0, t1, RZ * CZ);
  };

  // ---- SwiGLU transition (in place on xb) ----
  auto transition = [&](float* xb, const float* const* p, long long rows, int C) {
    int H = 4 * C;
    ln(xb, p[0], p[1], bfx, rows, C);
    cvt(p[2], w0, (long long)C * H);
    cvt(p[3], w1, (long long)C * H);
    cvt(p[4], w2, (long long)H * C);
    long long ck = (RZ * CZ) / H; if (ck > rows) ck = rows;
    ck &= ~127LL; if (ck < 128) ck = 128;          // keep M % 128 == 0
    for (long long r0 = 0; r0 < rows; r0 += ck) {
      long long cr = rows - r0; if (cr > ck) cr = ck;
      gemm2(bfx + r0 * C, w0, t0, (int)cr, H, C);
      gemm2(bfx + r0 * C, w1, t1, (int)cr, H, C);
      k_silu_mul<<<NB(cr * H), 256, 0, stream>>>(t0, t1, bf0, cr * H);
      gemm2(bf0, w2, t1, (int)cr, C, H);
      k_add<<<NB(cr * C), 256, 0, stream>>>(xb + r0 * C, t1, cr * C);
    }
  };

  // ---- AttentionPairBias (updates sb in place, reads zb) ----
  auto apb = [&](float* sb, const float* zb, const float* const* p) {
    ln(sb, p[0], p[1], sbf, N_, CS);
    cvt(p[2], w0, CS * CS); gemm2(sbf, w0, st0, N_, CS, CS);
    k_pad_head<<<NB(N_ * HS * 32), 256, 0, stream>>>(st0, p[3], qbf, 1);   // q (+bias), pad 24->32
    cvt(p[4], w0, CS * CS); gemm2(sbf, w0, st0, N_, CS, CS);
    k_pad_head<<<NB(N_ * HS * 32), 256, 0, stream>>>(st0, p[3], kbf, 0);   // k, pad 24->32
    cvt(p[5], w0, CS * CS); gemm2(sbf, w0, st0, N_, CS, CS); cvt(st0, vbf, (long long)N_ * CS);
    ln(zb, p[6], p[7], bfx, RZ, CZ);
    cvt(p[8], w0, CZ * HS); gemm2(bfx, w0, zbb, (int)RZ, HS, CZ);  // pair bias [N*N, 16]
    const float scale = 0.20412414523193154f;  // 1/sqrt(24)
    // scores[h][i][j], batch h = 16, padded head dim K=32
    gemm(qbf, kbf, scb, N_, N_, 32,
         HS * 32, 32, 0,
         HS * 32, 32, 0,
         N_, 1, (long long)N_ * N_, 0, HS, HS, 1, scale);
    int nrows = HS * N_;
    k_softmax_apb<<<(unsigned)((nrows + 7) / 8), 256, 0, stream>>>(scb, zbb, pbf, nrows);
    gemm(pbf, vbf, st0, N_, 24, N_,
         N_, (long long)N_ * N_, 0,
         CS, 24, 0,
         CS, 1, 24, 0, HS, HS, 0, 1.f);
    cvt(st0, obf, (long long)N_ * CS);
    cvt(p[10], w0, CS * CS); gemm2(obf, w0, st1, N_, CS, CS);      // out proj
    cvt(p[9],  w0, CS * CS); gemm2(sbf, w0, st0, N_, CS, CS);      // gate
    k_gate_residual<<<NB((long long)N_ * CS), 256, 0, stream>>>(sb, st0, st1, (long long)N_ * CS);
  };

  // ---- run the 2-block stack ----
  for (int blk = 0; blk < 2; ++blk) {
    const float* const* p = P + blk * 57;
    tri_mul(z, p + 0, true);      // tmo
    tri_mul(z, p + 10, false);    // tmi
    tri_att(z, p + 20);           // tas (starting node)
    k_transpose<<<NB(RZ * CZ), 256, 0, stream>>>(z, ztmp, N_, CZ);
    tri_att(ztmp, p + 28);        // tae (ending node on z^T)
    k_transpose<<<NB(RZ * CZ), 256, 0, stream>>>(ztmp, z, N_, CZ);
    transition(z, p + 36, RZ, CZ);        // pair transition
    apb(s, z, p + 41);                    // AttentionPairBias
    transition(s, p + 52, N_, CS);        // single transition
  }
}
